// TimeTokenEncoderBlock_59227599012603
// MI455X (gfx1250) — matrix operand order repaired and checked by reading, not compiled, not run on hardware
//
#include <hip/hip_runtime.h>
#include <hip/hip_bf16.h>
#include <math.h>

// ---- problem constants (match reference) ----
#define B_    32
#define L_    512
#define DM    512          // d_model
#define DT    256          // d_time
#define H_    8
#define HD_   64
#define DFF_  2048
#define NL_   2
#define NTOK  (B_ * L_)    // 16384 tokens
#define QKVW  (3 * DM)     // 1536
#define DCAT  (DM + DT)    // 768
#define EPS_  1e-5f

typedef __attribute__((ext_vector_type(16))) _Float16 v16h;
typedef __attribute__((ext_vector_type(8)))  float    v8f;
typedef _Float16 h16;

union Frag { v16h h; unsigned u[8]; };

// K index of low half of fragment register i (A-matrix 16x32 f16 layout)
__device__ __forceinline__ int a_kidx(int hi, int i) {
    return (i < 4) ? (hi * 8 + 2 * i) : (16 + hi * 8 + 2 * (i - 4));
}
// K index of low half of fragment register i (B-matrix 32x16 f16 layout)
__device__ __forceinline__ int b_kidx(int hi, int i) {
    return (i < 4) ? (hi * 8 + 2 * i) : (16 + hi * 8 + 2 * (i - 4));
}

// ------------------------------------------------------------------
// f32 -> f16 conversion (weights), 8 elements / thread, vectorized
// ------------------------------------------------------------------
__global__ void cvt_f32_f16_kernel(const float* __restrict__ src,
                                   h16* __restrict__ dst, int n) {
    int base = (blockIdx.x * blockDim.x + threadIdx.x) * 8;
    if (base < n) {
        float4 a = *(const float4*)&src[base];
        float4 b = *(const float4*)&src[base + 4];
        h16 o[8] = {(h16)a.x, (h16)a.y, (h16)a.z, (h16)a.w,
                    (h16)b.x, (h16)b.y, (h16)b.z, (h16)b.w};
        *(uint4*)&dst[base] = *(uint4*)o;
    }
}

// ------------------------------------------------------------------
// Tiled WMMA GEMM: C[M,N] = act(A16[M,K] @ W16[N,K]^T + bias)
// grid = (M/64, N/64), block = 256 (8 waves; each wave -> 16x32 tile)
// ------------------------------------------------------------------
__global__ __launch_bounds__(256)
void gemm_f16_kernel(const h16* __restrict__ A, const h16* __restrict__ W,
                     const float* __restrict__ bias,
                     float* __restrict__ C32, h16* __restrict__ C16,
                     int N, int K, int relu) {
    __shared__ __align__(16) h16 As[64][40];   // [m][k], 80B row stride
    __shared__ __align__(16) h16 Bs[64][40];   // [n][k]

    const int tid  = threadIdx.x;
    const int m0   = blockIdx.x * 64;
    const int n0   = blockIdx.y * 64;
    const int lane = tid & 31;
    const int wave = tid >> 5;
    const int wr   = wave >> 1;     // 0..3  (16-row tile)
    const int wc   = wave & 1;      // 0..1  (32-col tile)
    const int l    = lane & 15;
    const int hi   = lane >> 4;

    const int ar = tid >> 2;          // 0..63
    const int ac = (tid & 3) * 8;     // 0,8,16,24 (halves)

    v8f acc0 = {}, acc1 = {};
    for (int k0 = 0; k0 < K; k0 += 32) {
        // one 16B chunk per thread per tile (global_load_b128 -> ds_store_b128)
        *(uint4*)&As[ar][ac] = *(const uint4*)&A[(long)(m0 + ar) * K + k0 + ac];
        *(uint4*)&Bs[ar][ac] = *(const uint4*)&W[(long)(n0 + ar) * K + k0 + ac];
        if (k0 + 32 < K) {  // WGP-scope prefetch of next K tile
            __builtin_prefetch(&A[(long)(m0 + ar) * K + k0 + 32 + ac], 0, 3);
            __builtin_prefetch(&W[(long)(n0 + ar) * K + k0 + 32 + ac], 0, 3);
        }
        __syncthreads();

        Frag af, bf0, bf1;
        #pragma unroll
        for (int i = 0; i < 8; i++) {
            af.u[i]  = *(const unsigned*)&As[wr * 16 + l][a_kidx(hi, i)];
            int kb = b_kidx(hi, i);
            bf0.u[i] = *(const unsigned*)&Bs[wc * 32 + l][kb];
            bf1.u[i] = *(const unsigned*)&Bs[wc * 32 + 16 + l][kb];
        }
        acc0 = __builtin_amdgcn_wmma_f32_16x16x32_f16(false, af.h, false, bf0.h,
                                                      (short)0, acc0, false, false);
        acc1 = __builtin_amdgcn_wmma_f32_16x16x32_f16(false, af.h, false, bf1.h,
                                                      (short)0, acc1, false, false);
        __syncthreads();
    }

    const int rbase = m0 + wr * 16 + hi * 8;
    const int c0    = n0 + wc * 32 + l;
    const float bv0 = bias ? bias[c0] : 0.f;
    const float bv1 = bias ? bias[c0 + 16] : 0.f;
    #pragma unroll
    for (int r = 0; r < 8; r++) {
        float v0 = acc0[r] + bv0;
        float v1 = acc1[r] + bv1;
        if (relu) { v0 = fmaxf(v0, 0.f); v1 = fmaxf(v1, 0.f); }
        long o = (long)(rbase + r) * N + c0;
        if (C32) { C32[o] = v0; C32[o + 16] = v1; }
        if (C16) { C16[o] = (h16)v0; C16[o + 16] = (h16)v1; }
    }
}

// ------------------------------------------------------------------
// Row LayerNorm: out = LN(a[row, aoff:aoff+D] (+ resid[row])) * g + b
// optional f32 and f16 (strided/offset) outputs. One block per row.
// ------------------------------------------------------------------
__global__ __launch_bounds__(256)
void ln_kernel(const float* __restrict__ a, long lda, long aoff,
               const float* __restrict__ resid, long ldr,
               const float* __restrict__ g, const float* __restrict__ bi,
               float* __restrict__ out32, long ldo,
               h16* __restrict__ out16, long ld16, long off16, int D) {
    const int row = blockIdx.x;
    const int tid = threadIdx.x;
    const int nper = (D + 255) >> 8;
    float vals[4];
    float s = 0.f;
    for (int j = 0; j < nper; j++) {
        int d = tid + j * 256;
        float v = 0.f;
        if (d < D) {
            v = a[(long)row * lda + aoff + d];
            if (resid) v += resid[(long)row * ldr + d];
        }
        vals[j] = v;
        s += v;
    }
    __shared__ float red[256];
    red[tid] = s; __syncthreads();
    for (int st = 128; st > 0; st >>= 1) { if (tid < st) red[tid] += red[tid + st]; __syncthreads(); }
    float mean = red[0] / (float)D; __syncthreads();
    float s2 = 0.f;
    for (int j = 0; j < nper; j++) {
        int d = tid + j * 256;
        if (d < D) { float dv = vals[j] - mean; s2 += dv * dv; }
    }
    red[tid] = s2; __syncthreads();
    for (int st = 128; st > 0; st >>= 1) { if (tid < st) red[tid] += red[tid + st]; __syncthreads(); }
    float inv = rsqrtf(red[0] / (float)D + EPS_);
    for (int j = 0; j < nper; j++) {
        int d = tid + j * 256;
        if (d < D) {
            float y = (vals[j] - mean) * inv * g[d] + bi[d];
            if (out32) out32[(long)row * ldo + d] = y;
            if (out16) out16[(long)row * ld16 + off16 + d] = (h16)y;
        }
    }
}

// ------------------------------------------------------------------
// Time branch: x_time = LN((1 + diag[b,h,l]) * x_time_in) ; D=256.
// ------------------------------------------------------------------
__global__ __launch_bounds__(256)
void time_ln_kernel(const float* __restrict__ xt_in, const float* __restrict__ diag,
                    const float* __restrict__ g, const float* __restrict__ bi,
                    float* __restrict__ out32, h16* __restrict__ out16,
                    long ld16, long off16) {
    const int row = blockIdx.x;           // b*L + l
    const int b = row >> 9, lpos = row & 511;
    const int d = threadIdx.x;            // 0..255
    const int h = d >> 5;                 // head = d / 32
    float dg = diag[((long)b * H_ + h) * L_ + lpos];
    float v = xt_in[(long)row * DT + d] * (1.f + dg);

    __shared__ float red[256];
    red[d] = v; __syncthreads();
    for (int st = 128; st > 0; st >>= 1) { if (d < st) red[d] += red[d + st]; __syncthreads(); }
    float mean = red[0] * (1.f / 256.f); __syncthreads();
    float dv = v - mean;
    red[d] = dv * dv; __syncthreads();
    for (int st = 128; st > 0; st >>= 1) { if (d < st) red[d] += red[d + st]; __syncthreads(); }
    float inv = rsqrtf(red[0] * (1.f / 256.f) + EPS_);
    float y = dv * inv * g[d] + bi[d];
    out32[(long)row * DT + d] = y;
    out16[(long)row * ld16 + off16 + d] = (h16)y;
}

// ------------------------------------------------------------------
// Attention: per (qtile16, head, batch). 4 waves, cooperative staging.
// ------------------------------------------------------------------
__global__ __launch_bounds__(128)
void attn_kernel(const float* __restrict__ qkv, h16* __restrict__ tokattn,
                 float* __restrict__ diag) {
    __shared__ __align__(16) h16 sQ[16][72];        // q tile [q][dim]
    __shared__ __align__(16) h16 sKt[4][16][72];    // 4 K tiles [key][dim]
    __shared__ float sS[16][512];                   // scores / probs f32
    __shared__ __align__(16) h16 sP[16][520];       // probs f16 (padded rows)

    const int q0   = blockIdx.x * 16;
    const int h    = blockIdx.y;
    const int b    = blockIdx.z;
    const int tid  = threadIdx.x;
    const int lane = tid & 31;
    const int wave = tid >> 5;
    const int l    = lane & 15;
    const int hi   = lane >> 4;

    const float* base = qkv + (long)b * L_ * QKVW;
    const float* qb = base + h * HD_;
    const float* kb = base + DM + h * HD_;
    const float* vb = base + 2 * DM + h * HD_;

    // ---- stage Q tile (16x64): float4 reads, packed 4xf16 stores ----
    #pragma unroll
    for (int j = 0; j < 2; j++) {
        int idx = tid + j * 128;        // 0..255 float4 units
        int r = idx >> 4, c4 = idx & 15;
        float4 v = *(const float4*)&qb[(long)(q0 + r) * QKVW + c4 * 4];
        h16 p[4] = {(h16)v.x, (h16)v.y, (h16)v.z, (h16)v.w};
        *(uint2*)&sQ[r][c4 * 4] = *(uint2*)p;
    }
    __syncthreads();

    // ---- scores: S[16,512] = Q @ K^T / 8 ----
    for (int it = 0; it < 8; it++) {
        // cooperative stage of 4 K tiles (keys (it*4+w)*16..+15): float4 + b64 stores
        #pragma unroll
        for (int j = 0; j < 8; j++) {
            int idx = tid + j * 128;    // 0..1023 float4 units
            int w   = idx >> 8;
            int rem = idx & 255;
            int r   = rem >> 4;         // key row in tile
            int c4  = rem & 15;         // float4 column
            float4 v = *(const float4*)&kb[(long)((it * 4 + w) * 16 + r) * QKVW + c4 * 4];
            h16 p[4] = {(h16)v.x, (h16)v.y, (h16)v.z, (h16)v.w};
            *(uint2*)&sKt[w][r][c4 * 4] = *(uint2*)p;
        }
        __syncthreads();

        int kt = it * 4 + wave;
        v8f acc = {};
        #pragma unroll
        for (int kd0 = 0; kd0 < HD_; kd0 += 32) {
            Frag af, bf;
            #pragma unroll
            for (int i = 0; i < 8; i++) {
                af.u[i] = *(const unsigned*)&sQ[l][kd0 + a_kidx(hi, i)];
                bf.u[i] = *(const unsigned*)&sKt[wave][l][kd0 + b_kidx(hi, i)];
            }
            acc = __builtin_amdgcn_wmma_f32_16x16x32_f16(false, af.h, false, bf.h,
                                                         (short)0, acc, false, false);
        }
        #pragma unroll
        for (int r = 0; r < 8; r++)
            sS[hi * 8 + r][kt * 16 + l] = acc[r] * 0.125f;
        __syncthreads();
    }

    // ---- softmax over 512 keys; wave handles 4 rows ----
    for (int rr = 0; rr < 4; rr++) {
        int row = wave * 4 + rr;
        float m = -1e30f;
        for (int c = lane; c < L_; c += 32) m = fmaxf(m, sS[row][c]);
        #pragma unroll
        for (int s = 16; s > 0; s >>= 1) m = fmaxf(m, __shfl_xor(m, s, 32));
        float sum = 0.f;
        for (int c = lane; c < L_; c += 32) {
            float e = __expf(sS[row][c] - m);
            sS[row][c] = e;
            sum += e;
        }
        #pragma unroll
        for (int s = 16; s > 0; s >>= 1) sum += __shfl_xor(sum, s, 32);
        float inv = 1.f / sum;
        for (int c = lane; c < L_; c += 32) {
            float p = sS[row][c] * inv;
            sS[row][c] = p;
            sP[row][c] = (h16)p;
        }
        if (lane == 0)
            diag[((long)b * H_ + h) * L_ + q0 + row] = sS[row][q0 + row];
    }
    __syncthreads();

    // ---- O[16,64] = P @ V ; V staged transposed [dim][key], shared ----
    h16 (*sVt)[40] = (h16 (*)[40])&sKt[0][0][0];   // alias: [64 dims][40]
    v8f acc = {};
    for (int kt = 0; kt < 16; kt++) {
        // stage: key-pairs packed into b32 stores; dim-major coalesced reads
        #pragma unroll
        for (int j = 0; j < 8; j++) {
            int idx = tid + j * 128;    // 0..1023
            int dim = idx & 63;
            int kp  = idx >> 6;         // 0..15 (key pair)
            float a0 = vb[(long)(kt * 32 + kp * 2) * QKVW + dim];
            float a1 = vb[(long)(kt * 32 + kp * 2 + 1) * QKVW + dim];
            h16 p[2] = {(h16)a0, (h16)a1};
            *(unsigned*)&sVt[dim][kp * 2] = *(unsigned*)p;
        }
        __syncthreads();
        Frag af, bf;
        #pragma unroll
        for (int i = 0; i < 8; i++) {
            af.u[i] = *(const unsigned*)&sP[l][kt * 32 + a_kidx(hi, i)];
            bf.u[i] = *(const unsigned*)&sVt[wave * 16 + l][b_kidx(hi, i)];
        }
        acc = __builtin_amdgcn_wmma_f32_16x16x32_f16(false, af.h, false, bf.h,
                                                     (short)0, acc, false, false);
        __syncthreads();
    }
    #pragma unroll
    for (int r = 0; r < 8; r++)
        tokattn[(long)(b * L_ + q0 + hi * 8 + r) * DM + h * HD_ + wave * 16 + l] = (h16)acc[r];
}

// ------------------------------------------------------------------
// MaxPool over seq (kernel=stride=16) + concat -> pooled16 [B*32, 768]
// ------------------------------------------------------------------
__global__ void pool_kernel(const float* __restrict__ xtok,
                            const float* __restrict__ xtime,
                            h16* __restrict__ pooled) {
    int idx = blockIdx.x * blockDim.x + threadIdx.x;
    if (idx >= B_ * 32 * DCAT) return;
    int d = idx % DCAT;
    int bt = idx / DCAT;
    int b = bt >> 5, t = bt & 31;
    float m = -1e30f;
    if (d < DM) {
        for (int j = 0; j < 16; j++)
            m = fmaxf(m, xtok[((long)b * L_ + t * 16 + j) * DM + d]);
    } else {
        for (int j = 0; j < 16; j++)
            m = fmaxf(m, xtime[((long)b * L_ + t * 16 + j) * DT + (d - DM)]);
    }
    pooled[idx] = (h16)m;
}

// ------------------------------------------------------------------
// out = LN(gelu_exact(x)) ; rows of 256, one block per row.
// ------------------------------------------------------------------
__global__ __launch_bounds__(256)
void gelu_ln_kernel(const float* __restrict__ x, const float* __restrict__ g,
                    const float* __restrict__ bi, float* __restrict__ out) {
    const int row = blockIdx.x;
    const int d = threadIdx.x;
    float v = x[(long)row * 256 + d];
    v = 0.5f * v * (1.f + erff(v * 0.70710678118654752f));
    __shared__ float red[256];
    red[d] = v; __syncthreads();
    for (int st = 128; st > 0; st >>= 1) { if (d < st) red[d] += red[d + st]; __syncthreads(); }
    float mean = red[0] * (1.f / 256.f); __syncthreads();
    float dv = v - mean;
    red[d] = dv * dv; __syncthreads();
    for (int st = 128; st > 0; st >>= 1) { if (d < st) red[d] += red[d + st]; __syncthreads(); }
    float inv = rsqrtf(red[0] * (1.f / 256.f) + EPS_);
    out[(long)row * 256 + d] = dv * inv * g[d] + bi[d];
}

// ==================================================================
extern "C" void kernel_launch(void* const* d_in, const int* in_sizes, int n_in,
                              void* d_out, int out_size, void* d_ws, size_t ws_size,
                              hipStream_t stream) {
    (void)in_sizes; (void)n_in; (void)out_size; (void)ws_size;

    const float* src_token = (const float*)d_in[0];
    const float* src_time  = (const float*)d_in[1];
    const float* norm0_g   = (const float*)d_in[2];
    const float* norm0_b   = (const float*)d_in[3];
    const float* in_w      = (const float*)d_in[4];
    const float* in_b      = (const float*)d_in[5];
    const float* qkv_w     = (const float*)d_in[6];
    const float* qkv_b     = (const float*)d_in[7];
    const float* ao_w      = (const float*)d_in[8];
    const float* ao_b      = (const float*)d_in[9];
    const float* lin1_w    = (const float*)d_in[10];
    const float* lin1_b    = (const float*)d_in[11];
    const float* lin2_w    = (const float*)d_in[12];
    const float* lin2_b    = (const float*)d_in[13];
    const float* n1_g      = (const float*)d_in[14];
    const float* n1_b      = (const float*)d_in[15];
    const float* n2_g      = (const float*)d_in[16];
    const float* n2_b      = (const float*)d_in[17];
    const float* n3_g      = (const float*)d_in[18];
    const float* n3_b      = (const float*)d_in[19];
    const float* n4_g      = (const float*)d_in[20];
    const float* n4_b      = (const float*)d_in[21];
    const float* out_w     = (const float*)d_in[22];
    const float* out_b     = (const float*)d_in[23];
    const float* ln_g      = (const float*)d_in[24];
    const float* ln_b      = (const float*)d_in[25];
    float* out = (float*)d_out;

    // ---- deterministic workspace carve ----
    char* wsb = (char*)d_ws;
    size_t off = 0;
    auto carve = [&](size_t bytes) -> char* {
        char* p = wsb + off;
        off += (bytes + 255) & ~(size_t)255;
        return p;
    };
    const size_t N_INW  = (size_t)NL_ * DM * DM;
    const size_t N_QKVW = (size_t)NL_ * QKVW * DM;
    const size_t N_AOW  = (size_t)NL_ * DM * DM;
    const size_t N_L1W  = (size_t)NL_ * DFF_ * DCAT;
    const size_t N_L2W  = (size_t)NL_ * DCAT * DFF_;
    const size_t N_OUTW = (size_t)DT * DCAT;

    h16* w_in16  = (h16*)carve(N_INW  * 2);
    h16* w_qkv16 = (h16*)carve(N_QKVW * 2);
    h16* w_ao16  = (h16*)carve(N_AOW  * 2);
    h16* w_l116  = (h16*)carve(N_L1W  * 2);
    h16* w_l216  = (h16*)carve(N_L2W  * 2);
    h16* w_out16 = (h16*)carve(N_OUTW * 2);

    float* xtok   = (float*)carve((size_t)NTOK * DM * 4);
    float* xtime  = (float*)carve((size_t)NTOK * DT * 4);
    float* diag   = (float*)carve((size_t)B_ * H_ * L_ * 4);
    h16*   buf16a = (h16*)carve((size_t)NTOK * DFF_ * 2);   // A16 / tokattn16 / ff16
    h16*   buf16b = (h16*)carve((size_t)NTOK * DCAT * 2);   // xtok16 / concat16
    float* fbig   = (float*)carve((size_t)NTOK * QKVW * 4); // qkv / ao-out / ff2
    h16*   pooled = (h16*)carve((size_t)B_ * 32 * DCAT * 2);
    float* pout   = (float*)carve((size_t)B_ * 32 * DT * 4);

    // ---- convert weights to f16 ----
    auto cvt = [&](const float* s, h16* d, size_t n) {
        unsigned blocks = (unsigned)((n / 8 + 255) / 256);
        cvt_f32_f16_kernel<<<dim3(blocks), dim3(256), 0, stream>>>(s, d, (int)n);
    };
    cvt(in_w,   w_in16,  N_INW);
    cvt(qkv_w,  w_qkv16, N_QKVW);
    cvt(ao_w,   w_ao16,  N_AOW);
    cvt(lin1_w, w_l116,  N_L1W);
    cvt(lin2_w, w_l216,  N_L2W);
    cvt(out_w,  w_out16, N_OUTW);

    for (int i = 0; i < NL_; i++) {
        const float* tok_in  = (i == 0) ? src_token : xtok;
        const float* time_in = (i == 0) ? src_time  : xtime;

        // 1) norm0 LN -> f16 GEMM input (buf16a as [NTOK,512])
        ln_kernel<<<dim3(NTOK), dim3(256), 0, stream>>>(
            tok_in, DM, 0, nullptr, 0, norm0_g + i * DM, norm0_b + i * DM,
            nullptr, 0, buf16a, DM, 0, DM);

        // 2) in_proj GEMM 512->512 : writes xtok(f32) + xtok16(f16)
        gemm_f16_kernel<<<dim3(NTOK / 64, DM / 64), dim3(256), 0, stream>>>(
            buf16a, w_in16 + (size_t)i * DM * DM, in_b + i * DM,
            xtok, buf16b, DM, DM, 0);

        // 3) QKV GEMM 512->1536
        gemm_f16_kernel<<<dim3(NTOK / 64, QKVW / 64), dim3(256), 0, stream>>>(
            buf16b, w_qkv16 + (size_t)i * QKVW * DM, qkv_b + i * QKVW,
            fbig, nullptr, QKVW, DM, 0);

        // 4) attention -> tokattn16 (buf16a as [NTOK,512]) + diag
        attn_kernel<<<dim3(L_ / 16, H_, B_), dim3(128), 0, stream>>>(
            fbig, buf16a, diag);

        // 5) attn output projection 512->512 (overwrites fbig; qkv consumed)
        gemm_f16_kernel<<<dim3(NTOK / 64, DM / 64), dim3(256), 0, stream>>>(
            buf16a, w_ao16 + (size_t)i * DM * DM, ao_b + i * DM,
            fbig, nullptr, DM, DM, 0);

        // 6) n1: x_tok = LN(ao_out + x_tok); also f16 -> concat[:,0:512]
        ln_kernel<<<dim3(NTOK), dim3(256), 0, stream>>>(
            fbig, DM, 0, xtok, DM, n1_g + i * DM, n1_b + i * DM,
            xtok, DM, buf16b, DCAT, 0, DM);

        // 7) n2: x_time = LN((1+diag)*x_time); also f16 -> concat[:,512:768]
        time_ln_kernel<<<dim3(NTOK), dim3(256), 0, stream>>>(
            time_in, diag, n2_g + i * DT, n2_b + i * DT,
            xtime, buf16b, DCAT, DM);

        // 8) lin1 GEMM 768->2048 + ReLU, f16 out (buf16a as [NTOK,2048])
        gemm_f16_kernel<<<dim3(NTOK / 64, DFF_ / 64), dim3(256), 0, stream>>>(
            buf16b, w_l116 + (size_t)i * DFF_ * DCAT, lin1_b + i * DFF_,
            nullptr, buf16a, DFF_, DCAT, 1);

        // 9) lin2 GEMM 2048->768 (f32 into fbig as [NTOK,768])
        gemm_f16_kernel<<<dim3(NTOK / 64, DCAT / 64), dim3(256), 0, stream>>>(
            buf16a, w_l216 + (size_t)i * DCAT * DFF_, lin2_b + i * DCAT,
            fbig, nullptr, DCAT, DFF_, 0);

        // 10) n3: x_tok = LN(ff[:, :512] + x_tok)
        ln_kernel<<<dim3(NTOK), dim3(256), 0, stream>>>(
            fbig, DCAT, 0, xtok, DM, n3_g + i * DM, n3_b + i * DM,
            xtok, DM, nullptr, 0, 0, DM);

        // 11) n4: x_time = LN(ff[:, 512:] + x_time)
        ln_kernel<<<dim3(NTOK), dim3(256), 0, stream>>>(
            fbig, DCAT, DM, xtime, DT, n4_g + i * DT, n4_b + i * DT,
            xtime, DT, nullptr, 0, 0, DT);
    }

    // pool (max over seq/16) + concat -> pooled16 [1024, 768]
    {
        int n = B_ * 32 * DCAT;
        pool_kernel<<<dim3((n + 255) / 256), dim3(256), 0, stream>>>(xtok, xtime, pooled);
    }
    // out projection 768->256
    gemm_f16_kernel<<<dim3((B_ * 32) / 64, DT / 64), dim3(256), 0, stream>>>(
        pooled, w_out16, out_b, pout, nullptr, DT, DCAT, 0);
    // exact GELU + final LN -> d_out
    gelu_ln_kernel<<<dim3(B_ * 32), dim3(256), 0, stream>>>(pout, ln_g, ln_b, out);
}
